// DualModalityEnhanced_62251255988301
// MI455X (gfx1250) — matrix operand order repaired and checked by reading, not compiled, not run on hardware
//
#include <hip/hip_runtime.h>
#include <hip/hip_bf16.h>

typedef __attribute__((ext_vector_type(16))) _Float16 v16h;
typedef __attribute__((ext_vector_type(8)))  _Float16 v8h;
typedef __attribute__((ext_vector_type(4)))  _Float16 v4h;
typedef __attribute__((ext_vector_type(8)))  float    v8f;
typedef int v4i_g __attribute__((vector_size(16)));   // matches builtin param type

#define KT 32   // K tile (one WMMA K-step)
#define KS 48   // LDS row stride in halves (96B -> 16B/32B aligned fragment loads)

// ---------------------------------------------------------------------------
// 16-byte async copy global -> LDS (ASYNCcnt-tracked, bypasses VGPRs).
// Builtin signature (from probe): (AS1 v4i* src, AS3 v4i* lds_dst, imm, imm cpol)
// ---------------------------------------------------------------------------
__device__ __forceinline__ void async_copy16(_Float16* lds_dst, const _Float16* gsrc)
{
#if __has_builtin(__builtin_amdgcn_global_load_async_to_lds_b128)
    typedef __attribute__((address_space(1))) v4i_g* gp_t;
    typedef __attribute__((address_space(3))) v4i_g* sp_t;
    __builtin_amdgcn_global_load_async_to_lds_b128(
        (gp_t)(unsigned long long)gsrc,
        (sp_t)(unsigned long long)lds_dst, 0, 0);
#else
    asm volatile("global_load_async_to_lds_b128 %0, %1, off"
                 :: "v"((unsigned)(unsigned long long)lds_dst),
                    "v"((unsigned long long)gsrc)
                 : "memory");
#endif
}

template<int N>
__device__ __forceinline__ void wait_async()
{
#if __has_builtin(__builtin_amdgcn_s_wait_asynccnt)
    __builtin_amdgcn_s_wait_asynccnt((unsigned short)N);
#else
    asm volatile("s_wait_asynccnt %0" :: "i"(N) : "memory");
#endif
}

// ---------------------------------------------------------------------------
// f16-in / f32-accumulate GEMM via v_wmma_f32_16x16x32_f16.
//   acc = A[z][m][k] * B^T   (BT=true : B stored [z][n][k]; async double-buffered)
//                            (BT=false: B stored [z][k][n]; sync LDS transpose)
//   C   (f32, optional) = acc + bias (+R) (+relu)
//   Ch  (f16, optional) = same value, half precision mirror for next consumer
// Wave tile = (TM*16) x (TN*16); block = WM x WN waves.
// All dims are exact multiples of the tile sizes for the shapes used here.
// ---------------------------------------------------------------------------
template<int WM, int WN, int TM, int TN, bool BT>
__global__ __launch_bounds__(WM*WN*32)
void gemm_f16_wmma_kernel(const _Float16* __restrict__ A, int lda, long long az,
                          const _Float16* __restrict__ Bm, int ldb, long long bz,
                          const float* __restrict__ bias,
                          const float* __restrict__ R,
                          float* __restrict__ C,
                          _Float16* __restrict__ Ch,
                          int ldc, long long cz,
                          int K, int relu)
{
    constexpr int BM   = WM * TM * 16;
    constexpr int BN   = WN * TN * 16;
    constexpr int NT   = WM * WN * 32;
    constexpr int NBUF = BT ? 2 : 1;
    constexpr int AU   = (BM * 4) / NT;   // 8-half units per thread (A)
    constexpr int BU   = (BN * 4) / NT;   // 8-half units per thread (B)
    static_assert((BM * 4) % NT == 0 && (BN * 4) % NT == 0, "staging divisibility");

    __shared__ _Float16 Al[NBUF][BM][KS];
    __shared__ _Float16 Bl[NBUF][BN][KS];   // always [n][k]

    const int z = blockIdx.z;
    const _Float16* Ab = A  + (long long)z * az;
    const _Float16* Bb = Bm + (long long)z * bz;

    const int m0   = blockIdx.y * BM;
    const int n0   = blockIdx.x * BN;
    const int tid  = threadIdx.x;
    const int lane = tid & 31;
    const int wave = tid >> 5;
    const int mw   = (wave / WN) * (TM * 16);
    const int nw   = (wave % WN) * (TN * 16);

    v8f acc[TM][TN] = {};

    const int l15 = lane & 15;
    const int kb  = (lane >> 4) * 8;    // unused; A and B both read 16 contiguous K at kh
    const int kh  = (lane >> 4) * 16;   // B: contiguous 16 K per half-wave

    auto compute_tile = [&](int bsel) {
        v16h af[TM], bf[TN];
        #pragma unroll
        for (int i = 0; i < TM; ++i) {
            const _Float16* ap = &Al[bsel][mw + i * 16 + l15][0];
            v8h lo = *(const v8h*)(ap + kh);
            v8h hi = *(const v8h*)(ap + kh + 8);
            af[i] = __builtin_shufflevector(lo, hi, 0,1,2,3,4,5,6,7,8,9,10,11,12,13,14,15);
        }
        #pragma unroll
        for (int j = 0; j < TN; ++j)
            bf[j] = *(const v16h*)&Bl[bsel][nw + j * 16 + l15][kh];
        #pragma unroll
        for (int i = 0; i < TM; ++i)
            #pragma unroll
            for (int j = 0; j < TN; ++j)
                acc[i][j] = __builtin_amdgcn_wmma_f32_16x16x32_f16(
                    false, af[i], false, bf[j], (short)0, acc[i][j], false, false);
    };

    if constexpr (BT) {
        // ---- async double-buffered pipeline: copy tile t+1 overlaps WMMA of t
        auto issue = [&](int k0, int bsel) {
            #pragma unroll
            for (int it = 0; it < AU; ++it) {
                const int u = tid + it * NT;
                const int r = u >> 2, c = (u & 3) * 8;
                async_copy16(&Al[bsel][r][c], Ab + (long long)(m0 + r) * lda + (k0 + c));
            }
            #pragma unroll
            for (int it = 0; it < BU; ++it) {
                const int u = tid + it * NT;
                const int r = u >> 2, c = (u & 3) * 8;
                async_copy16(&Bl[bsel][r][c], Bb + (long long)(n0 + r) * ldb + (k0 + c));
            }
        };
        const int T = K / KT;
        issue(0, 0);
        for (int t = 0; t < T; ++t) {
            if (t + 1 < T) { issue((t + 1) * KT, (t + 1) & 1); wait_async<AU + BU>(); }
            else          { wait_async<0>(); }
            __syncthreads();
            compute_tile(t & 1);
            __syncthreads();
        }
    } else {
        // ---- synchronous single-buffer path (B needs LDS transpose-scatter)
        for (int k0 = 0; k0 < K; k0 += KT) {
            #pragma unroll
            for (int it = 0; it < AU; ++it) {
                const int u = tid + it * NT;
                const int r = u >> 2, c = (u & 3) * 8;
                const _Float16* src = Ab + (long long)(m0 + r) * lda + (k0 + c);
                *(v8h*)&Al[0][r][c] = *(const v8h*)src;
                if (it == 0) __builtin_prefetch(src + KT, 0, 1);
            }
            #pragma unroll
            for (int it = 0; it < BU; ++it) {
                const int u = tid + it * NT;
                const int k = u / (BN / 8);
                const int n = (u % (BN / 8)) * 8;
                const _Float16* src = Bb + (long long)(k0 + k) * ldb + (n0 + n);
                v8h vv = *(const v8h*)src;
                if (it == 0) __builtin_prefetch(src + (long long)KT * ldb, 0, 1);
                #pragma unroll
                for (int e = 0; e < 8; ++e) Bl[0][n + e][k] = vv[e];
            }
            __syncthreads();
            compute_tile(0);
            __syncthreads();
        }
    }

    // ---- epilogue: lane = N col, VGPR v = M row (+8 for upper half-wave) ----
    float*      Cb  = C  ? C  + (long long)z * cz : nullptr;
    _Float16*   Chb = Ch ? Ch + (long long)z * cz : nullptr;
    const int col   = lane & 15;
    const int rbase = (lane >> 4) * 8;
    #pragma unroll
    for (int i = 0; i < TM; ++i) {
        #pragma unroll
        for (int j = 0; j < TN; ++j) {
            const int gn = n0 + nw + j * 16 + col;
            const float bv = bias ? bias[gn] : 0.0f;
            #pragma unroll
            for (int v = 0; v < 8; ++v) {
                const int gm = m0 + mw + i * 16 + rbase + v;
                float x = acc[i][j][v] + bv;
                if (R)    x += R[(long long)gm * ldc + gn];
                if (relu) x = fmaxf(x, 0.0f);
                const long long o = (long long)gm * ldc + gn;
                if (Cb)  Cb[o]  = x;
                if (Chb) Chb[o] = (_Float16)x;
            }
        }
    }
}

// ---------------------------------------------------------------------------
// p = softmax(scale * x) over 512 cols; f32 in, f16 out. One wave per row.
// ---------------------------------------------------------------------------
__global__ __launch_bounds__(256)
void softmax512_kernel(const float* __restrict__ P, _Float16* __restrict__ O,
                       int rows, float scale)
{
    const int wave = threadIdx.x >> 5;
    const int lane = threadIdx.x & 31;
    const int row  = blockIdx.x * 8 + wave;
    if (row >= rows) return;
    const float* p = P + (long long)row * 512;
    _Float16*    o = O + (long long)row * 512;

    float v[16];
    float mx = -3.4e38f;
    #pragma unroll
    for (int j = 0; j < 16; ++j) { v[j] = p[j * 32 + lane]; mx = fmaxf(mx, v[j]); }
    #pragma unroll
    for (int t = 16; t >= 1; t >>= 1) mx = fmaxf(mx, __shfl_xor(mx, t, 32));
    float s = 0.0f;
    #pragma unroll
    for (int j = 0; j < 16; ++j) { v[j] = __expf(scale * (v[j] - mx)); s += v[j]; }
    #pragma unroll
    for (int t = 16; t >= 1; t >>= 1) s += __shfl_xor(s, t, 32);
    const float inv = 1.0f / s;
    #pragma unroll
    for (int j = 0; j < 16; ++j) o[j * 32 + lane] = (_Float16)(v[j] * inv);
}

// ---------------------------------------------------------------------------
// LayerNorm over 1024 cols; f32 in, f16 out. One block (256 thr) per row.
// ---------------------------------------------------------------------------
__global__ __launch_bounds__(256)
void layernorm1024_kernel(const float* __restrict__ X,
                          const float* __restrict__ g,
                          const float* __restrict__ b,
                          _Float16* __restrict__ Y)
{
    const int row = blockIdx.x;
    const int tid = threadIdx.x;
    const float* x = X + (long long)row * 1024;

    float vv[4];
    float s = 0.0f, ss = 0.0f;
    #pragma unroll
    for (int j = 0; j < 4; ++j) {
        float t = x[tid + j * 256];
        vv[j] = t; s += t; ss += t * t;
    }
    #pragma unroll
    for (int t = 16; t >= 1; t >>= 1) { s += __shfl_xor(s, t, 32); ss += __shfl_xor(ss, t, 32); }

    __shared__ float rs[8], rss[8];
    const int wave = tid >> 5, lane = tid & 31;
    if (lane == 0) { rs[wave] = s; rss[wave] = ss; }
    __syncthreads();
    if (tid == 0) {
        float S = 0.0f, SS = 0.0f;
        for (int i = 0; i < 8; ++i) { S += rs[i]; SS += rss[i]; }
        rs[0] = S; rss[0] = SS;
    }
    __syncthreads();
    const float mean = rs[0]  * (1.0f / 1024.0f);
    const float var  = rss[0] * (1.0f / 1024.0f) - mean * mean;
    const float rv   = rsqrtf(var + 1e-5f);
    #pragma unroll
    for (int j = 0; j < 4; ++j) {
        const int c = tid + j * 256;
        Y[(long long)row * 1024 + c] = (_Float16)((vv[j] - mean) * rv * g[c] + b[c]);
    }
}

// ---------------------------------------------------------------------------
// Wt[n][k] (f16) = W[k][n] (f32). 32x32 LDS tile transpose + convert.
// ---------------------------------------------------------------------------
__global__ __launch_bounds__(256)
void transpose_cvt_kernel(const float* __restrict__ W, _Float16* __restrict__ Wt,
                          int Kd, int Nd)
{
    __shared__ _Float16 t[32][33];
    const int k0 = blockIdx.y * 32, n0 = blockIdx.x * 32;
    const int tid = threadIdx.x;
    const int r = tid >> 3, c = (tid & 7) * 4;
    float4 f = *(const float4*)(W + (long long)(k0 + r) * Nd + (n0 + c));
    t[r][c + 0] = (_Float16)f.x;  t[r][c + 1] = (_Float16)f.y;
    t[r][c + 2] = (_Float16)f.z;  t[r][c + 3] = (_Float16)f.w;
    __syncthreads();
    v4h o = { t[c + 0][r], t[c + 1][r], t[c + 2][r], t[c + 3][r] };
    *(v4h*)(Wt + (long long)(n0 + r) * Kd + (k0 + c)) = o;
}

// ---------------------------------------------------------------------------
// outf (f32) and outh (f16) copies of in (f32). n must be multiple of 1024.
// ---------------------------------------------------------------------------
__global__ __launch_bounds__(256)
void dupcvt_kernel(const float* __restrict__ in, float* __restrict__ outf,
                   _Float16* __restrict__ outh)
{
    const int i = blockIdx.x * 256 + threadIdx.x;
    float4 f = ((const float4*)in)[i];
    ((float4*)outf)[i] = f;
    v4h h = { (_Float16)f.x, (_Float16)f.y, (_Float16)f.z, (_Float16)f.w };
    ((v4h*)outh)[i] = h;
}

// ---------------------------------------------------------------------------
template<int WM, int WN, int TM, int TN, bool BT>
static inline void launch_gemm(hipStream_t st, int M, int N, int K,
                               const _Float16* A, int lda, long long az,
                               const _Float16* Bm, int ldb, long long bz,
                               const float* bias, const float* R,
                               float* C, _Float16* Ch, int ldc, long long cz,
                               int Z, int relu)
{
    dim3 grid(N / (WN * TN * 16), M / (WM * TM * 16), Z);
    dim3 block(WM * WN * 32);
    gemm_f16_wmma_kernel<WM, WN, TM, TN, BT><<<grid, block, 0, st>>>(
        A, lda, az, Bm, ldb, bz, bias, R, C, Ch, ldc, cz, K, relu);
}

extern "C" void kernel_launch(void* const* d_in, const int* in_sizes, int n_in,
                              void* d_out, int out_size, void* d_ws, size_t ws_size,
                              hipStream_t stream)
{
    (void)in_sizes; (void)n_in; (void)out_size; (void)ws_size;
    const int B = 4, S = 512, D = 1024, H = 16, HD = 64, HID = 4096, NL = 2;
    const int M = B * S;                       // 2048 rows
    const float scale = 0.125f;                // 1/sqrt(64)
    const size_t MD = (size_t)M * D;
    const size_t SS = (size_t)H * S * S;

    const float* feat[3] = {(const float*)d_in[0], (const float*)d_in[1], (const float*)d_in[2]};
    const float* WQ  = (const float*)d_in[3];  const float* bQ  = (const float*)d_in[4];
    const float* WK  = (const float*)d_in[5];  const float* bK  = (const float*)d_in[6];
    const float* WV1 = (const float*)d_in[7];  const float* bV1 = (const float*)d_in[8];
    const float* WV2 = (const float*)d_in[9];  const float* bV2 = (const float*)d_in[10];
    const float* WO1 = (const float*)d_in[11]; const float* bO1 = (const float*)d_in[12];
    const float* WO2 = (const float*)d_in[13]; const float* bO2 = (const float*)d_in[14];
    const float* F1W1= (const float*)d_in[15]; const float* F1b1= (const float*)d_in[16];
    const float* F1W2= (const float*)d_in[17]; const float* F1b2= (const float*)d_in[18];
    const float* F2W1= (const float*)d_in[19]; const float* F2b1= (const float*)d_in[20];
    const float* F2W2= (const float*)d_in[21]; const float* F2b2= (const float*)d_in[22];
    const float* LN1g= (const float*)d_in[23]; const float* LN1b= (const float*)d_in[24];
    const float* LN2g= (const float*)d_in[25]; const float* LN2b= (const float*)d_in[26];

    // ---- workspace carve-out (~200 MB) ----
    char* base = (char*)d_ws;
    size_t off = 0;
    auto carve = [&](size_t bytes) { void* p = base + off; off = (off + bytes + 255) & ~(size_t)255; return p; };
    float* o1  = (float*)carve(MD * 4);  float* o2 = (float*)carve(MD * 4);
    float* m1  = (float*)carve(MD * 4);  float* m2 = (float*)carve(MD * 4);
    float* S1b = (float*)carve(SS * 4);  float* S2b = (float*)carve(SS * 4);
    _Float16* m1h = (_Float16*)carve(MD * 2);  _Float16* m2h = (_Float16*)carve(MD * 2);
    _Float16* Qh  = (_Float16*)carve(MD * 2);  _Float16* Kh  = (_Float16*)carve(MD * 2);
    _Float16* V1h = (_Float16*)carve(MD * 2);  _Float16* V2h = (_Float16*)carve(MD * 2);
    _Float16* nV1h= (_Float16*)carve(MD * 2);  _Float16* nV2h= (_Float16*)carve(MD * 2);
    _Float16* x1h = (_Float16*)carve(MD * 2);  _Float16* x2h = (_Float16*)carve(MD * 2);
    _Float16* P1h = (_Float16*)carve(SS * 2);  _Float16* P2h = (_Float16*)carve(SS * 2);
    _Float16* h1h = (_Float16*)carve((size_t)M * HID * 2);
    _Float16* h2h = (_Float16*)carve((size_t)M * HID * 2);
    _Float16* QWt  = (_Float16*)carve((size_t)D * D * 2);
    _Float16* KWt  = (_Float16*)carve((size_t)D * D * 2);
    _Float16* V1Wt = (_Float16*)carve((size_t)D * D * 2);
    _Float16* V2Wt = (_Float16*)carve((size_t)D * D * 2);
    _Float16* O1Wt = (_Float16*)carve((size_t)D * D * 2);
    _Float16* O2Wt = (_Float16*)carve((size_t)D * D * 2);
    _Float16* FA1t = (_Float16*)carve((size_t)D * HID * 2);  // F1W1^T [HID][D]
    _Float16* FA2t = (_Float16*)carve((size_t)D * HID * 2);  // F2W1^T
    _Float16* FB1t = (_Float16*)carve((size_t)D * HID * 2);  // F1W2^T [D][HID]
    _Float16* FB2t = (_Float16*)carve((size_t)D * HID * 2);  // F2W2^T

    auto xpose = [&](const float* W, _Float16* Wt, int Kd, int Nd) {
        transpose_cvt_kernel<<<dim3(Nd / 32, Kd / 32), 256, 0, stream>>>(W, Wt, Kd, Nd);
    };

    for (int br = 0; br < 3; ++br) {
        const float* in1 = feat[br == 2 ? 1 : 0];  // TA:text TV:text AV:audio
        const float* in2 = feat[br == 0 ? 1 : 2];  // TA:audio TV:visual AV:visual
        dupcvt_kernel<<<MD / 1024, 256, 0, stream>>>(in1, m1, m1h);
        dupcvt_kernel<<<MD / 1024, 256, 0, stream>>>(in2, m2, m2h);

        for (int l = 0; l < NL; ++l) {
            const int i = br * NL + l;
            const size_t wo = (size_t)i * D * D, bo = (size_t)i * D;
            const size_t fw1 = (size_t)i * D * HID, fb1 = (size_t)i * HID;
            const size_t fw2 = (size_t)i * HID * D;

            // stage this layer's weights as transposed f16
            xpose(WQ  + wo, QWt,  D, D);   xpose(WK  + wo, KWt,  D, D);
            xpose(WV1 + wo, V1Wt, D, D);   xpose(WV2 + wo, V2Wt, D, D);
            xpose(WO1 + wo, O1Wt, D, D);   xpose(WO2 + wo, O2Wt, D, D);
            xpose(F1W1 + fw1, FA1t, D, HID);  xpose(F2W1 + fw1, FA2t, D, HID);
            xpose(F1W2 + fw2, FB1t, HID, D);  xpose(F2W2 + fw2, FB2t, HID, D);

            // projections: f16 out only (consumed as f16 by attention)
            launch_gemm<4,2,2,4,true>(stream, M, D, D, m1h, D, 0, QWt,  D, 0, bQ  + bo, nullptr, nullptr, Qh,  D, 0, 1, 0);
            launch_gemm<4,2,2,4,true>(stream, M, D, D, m2h, D, 0, KWt,  D, 0, bK  + bo, nullptr, nullptr, Kh,  D, 0, 1, 0);
            launch_gemm<4,2,2,4,true>(stream, M, D, D, m1h, D, 0, V1Wt, D, 0, bV1 + bo, nullptr, nullptr, V1h, D, 0, 1, 0);
            launch_gemm<4,2,2,4,true>(stream, M, D, D, m2h, D, 0, V2Wt, D, 0, bV2 + bo, nullptr, nullptr, V2h, D, 0, 1, 0);

            // attention, per batch element; heads batched via grid.z
            for (int b = 0; b < B; ++b) {
                const _Float16* Qp  = Qh  + (size_t)b * S * D;
                const _Float16* Kp  = Kh  + (size_t)b * S * D;
                const _Float16* V1p = V1h + (size_t)b * S * D;
                const _Float16* V2p = V2h + (size_t)b * S * D;

                // S1 = Q K^T ; S2 = K Q^T  (f32 scores, K=HD, A*B^T natural layout)
                launch_gemm<4,2,2,4,true>(stream, S, S, HD, Qp, D, HD, Kp, D, HD,
                                          nullptr, nullptr, S1b, nullptr, S, (long long)S * S, H, 0);
                launch_gemm<4,2,2,4,true>(stream, S, S, HD, Kp, D, HD, Qp, D, HD,
                                          nullptr, nullptr, S2b, nullptr, S, (long long)S * S, H, 0);
                softmax512_kernel<<<(H * S) / 8, 256, 0, stream>>>(S1b, P1h, H * S, scale);
                softmax512_kernel<<<(H * S) / 8, 256, 0, stream>>>(S2b, P2h, H * S, scale);

                // nV2 = P1 @ V2 ; nV1 = P2 @ V1  (f16 out; B row-major -> BT=false)
                launch_gemm<4,2,2,2,false>(stream, S, HD, S, P1h, S, (long long)S * S,
                                           V2p, D, HD, nullptr, nullptr, nullptr,
                                           nV2h + (size_t)b * S * D, D, HD, H, 0);
                launch_gemm<4,2,2,2,false>(stream, S, HD, S, P2h, S, (long long)S * S,
                                           V1p, D, HD, nullptr, nullptr, nullptr,
                                           nV1h + (size_t)b * S * D, D, HD, H, 0);
            }

            // output proj + residual (f32 out for LN/residual)
            launch_gemm<4,2,2,4,true>(stream, M, D, D, nV1h, D, 0, O1Wt, D, 0, bO1 + bo, m1, o1, nullptr, D, 0, 1, 0);
            launch_gemm<4,2,2,4,true>(stream, M, D, D, nV2h, D, 0, O2Wt, D, 0, bO2 + bo, m2, o2, nullptr, D, 0, 1, 0);

            // layernorm -> f16
            layernorm1024_kernel<<<M, 256, 0, stream>>>(o1, LN1g + bo, LN1b + bo, x1h);
            layernorm1024_kernel<<<M, 256, 0, stream>>>(o2, LN2g + bo, LN2b + bo, x2h);

            // FFN: h = relu(x W1 + b1) (f16 only); m_next = h W2 + b2 + o (f32 + f16)
            launch_gemm<4,2,2,4,true>(stream, M, HID, D, x1h, D, 0, FA1t, D, 0, F1b1 + fb1, nullptr, nullptr, h1h, HID, 0, 1, 1);
            launch_gemm<4,2,2,4,true>(stream, M, HID, D, x2h, D, 0, FA2t, D, 0, F2b1 + fb1, nullptr, nullptr, h2h, HID, 0, 1, 1);
            launch_gemm<4,2,2,4,true>(stream, M, D, HID, h1h, HID, 0, FB1t, HID, 0, F1b2 + bo, o1, m1, m1h, D, 0, 1, 0);
            launch_gemm<4,2,2,4,true>(stream, M, D, HID, h2h, HID, 0, FB2t, HID, 0, F2b2 + bo, o2, m2, m2h, D, 0, 1, 0);
        }

        (void)hipMemcpyAsync((float*)d_out + (size_t)(2 * br + 0) * MD, m1, MD * sizeof(float),
                             hipMemcpyDeviceToDevice, stream);
        (void)hipMemcpyAsync((float*)d_out + (size_t)(2 * br + 1) * MD, m2, MD * sizeof(float),
                             hipMemcpyDeviceToDevice, stream);
    }
}
